// MACoAttention_81578608820837
// MI455X (gfx1250) — matrix operand order repaired and checked by reading, not compiled, not run on hardware
//
#include <hip/hip_runtime.h>
#include <hip/hip_bf16.h>

// Problem constants (from reference): B=64, S=512, D=512, K=256
#define Bn 64
#define Sn 512
#define Dn 512
#define Kn 256

typedef __attribute__((ext_vector_type(16))) __bf16 v16bf;
typedef __attribute__((ext_vector_type(8)))  float  v8f;

// ---------------------------------------------------------------------------
// Helpers: convert contiguous f32 runs to the WMMA bf16 operand packing.
// ---------------------------------------------------------------------------

// 16 contiguous f32 -> v16bf (B-matrix per-lane payload: K=d0..d0+15 for the
// lane's half; elements 0..15 map to VGPR0..7, 2 per VGPR).
__device__ inline v16bf cvt_bf16_16(const float* __restrict__ p) {
  const float4* q = (const float4*)p;
  float4 f0 = q[0], f1 = q[1], f2 = q[2], f3 = q[3];
  v16bf r;
  r[0]  = (__bf16)f0.x; r[1]  = (__bf16)f0.y; r[2]  = (__bf16)f0.z; r[3]  = (__bf16)f0.w;
  r[4]  = (__bf16)f1.x; r[5]  = (__bf16)f1.y; r[6]  = (__bf16)f1.z; r[7]  = (__bf16)f1.w;
  r[8]  = (__bf16)f2.x; r[9]  = (__bf16)f2.y; r[10] = (__bf16)f2.z; r[11] = (__bf16)f2.w;
  r[12] = (__bf16)f3.x; r[13] = (__bf16)f3.y; r[14] = (__bf16)f3.z; r[15] = (__bf16)f3.w;
  return r;
}

// Two runs of 8 contiguous f32 -> v16bf (A-matrix per-lane payload:
// elements 0..7 = first run (VGPR0..3), elements 8..15 = second run (VGPR4..7)).
__device__ inline v16bf cvt_bf16_8_8(const float* __restrict__ p0,
                                     const float* __restrict__ p1) {
  const float4* a = (const float4*)p0;
  const float4* b = (const float4*)p1;
  float4 f0 = a[0], f1 = a[1], f2 = b[0], f3 = b[1];
  v16bf r;
  r[0]  = (__bf16)f0.x; r[1]  = (__bf16)f0.y; r[2]  = (__bf16)f0.z; r[3]  = (__bf16)f0.w;
  r[4]  = (__bf16)f1.x; r[5]  = (__bf16)f1.y; r[6]  = (__bf16)f1.z; r[7]  = (__bf16)f1.w;
  r[8]  = (__bf16)f2.x; r[9]  = (__bf16)f2.y; r[10] = (__bf16)f2.z; r[11] = (__bf16)f2.w;
  r[12] = (__bf16)f3.x; r[13] = (__bf16)f3.y; r[14] = (__bf16)f3.z; r[15] = (__bf16)f3.w;
  return r;
}

// ---------------------------------------------------------------------------
// Kernel 1: mean-pool both tensors over S and fuse M0 = tanh(V0 * Q0).
// grid = B, block = 256. Thread t owns columns d = t and t+256 (coalesced).
// ---------------------------------------------------------------------------
__global__ __launch_bounds__(256) void mean_tanh_kernel(
    const float* __restrict__ Vn, const float* __restrict__ Qn,
    float* __restrict__ M0) {
  const int b = blockIdx.x;
  const int t = threadIdx.x;
  const float* vb = Vn + (size_t)b * Sn * Dn;
  const float* qb = Qn + (size_t)b * Sn * Dn;
  float av0 = 0.f, av1 = 0.f, aq0 = 0.f, aq1 = 0.f;
  for (int s = 0; s < Sn; ++s) {
    const size_t r = (size_t)s * Dn;
    av0 += vb[r + t];       av1 += vb[r + t + 256];
    aq0 += qb[r + t];       aq1 += qb[r + t + 256];
  }
  const float inv = 1.0f / (float)Sn;
  M0[(size_t)b * Dn + t]       = tanhf((av0 * inv) * (aq0 * inv));
  M0[(size_t)b * Dn + t + 256] = tanhf((av1 * inv) * (aq1 * inv));
}

// ---------------------------------------------------------------------------
// Kernel 2: wg[b,k] = W_h[k] * tanh( W_m[k,:] . M0[b,:] )
// grid = B, block = 256 (one thread per k). M0 row staged in LDS.
// ---------------------------------------------------------------------------
__global__ __launch_bounds__(256) void gate_kernel(
    const float* __restrict__ Wm, const float* __restrict__ Wh,
    const float* __restrict__ M0, float* __restrict__ wg) {
  const int b = blockIdx.x;
  const int k = threadIdx.x;
  __shared__ float m0s[Dn];
  m0s[k]       = M0[(size_t)b * Dn + k];
  m0s[k + 256] = M0[(size_t)b * Dn + k + 256];
  __syncthreads();
  const float* wr = Wm + (size_t)k * Dn;
  float acc = 0.f;
  #pragma unroll 4
  for (int d = 0; d < Dn; ++d) acc += wr[d] * m0s[d];
  wg[(size_t)b * Kn + k] = Wh[k] * tanhf(acc);
}

// ---------------------------------------------------------------------------
// Kernel 3 (WMMA): scores[which][b][s] = sum_k wg[b,k]*tanh( (W X^T)[k,s] )
// grid = (B, S/16, 2), block = 128 (4 waves). Each wave owns 4 k-tiles and
// accumulates the full D=512 reduction with v_wmma_f32_16x16x32_bf16, then
// fuses tanh + wg weighting + k-reduction on the accumulators.
// ---------------------------------------------------------------------------
__global__ __launch_bounds__(128) void scores_kernel(
    const float* __restrict__ Vn, const float* __restrict__ Qn,
    const float* __restrict__ Wv, const float* __restrict__ Wq,
    const float* __restrict__ wg, float* __restrict__ scores) {
  const int b     = blockIdx.x;
  const int s0    = blockIdx.y * 16;
  const int which = blockIdx.z;
  const int lane  = threadIdx.x & 31;
  const int wave  = threadIdx.x >> 5;
  const int half  = lane >> 4;      // 0: lanes 0-15, 1: lanes 16-31
  const int l16   = lane & 15;

  const float* X = which ? Qn : Vn;
  const float* W = which ? Wq : Wv;

  // B-operand source row: lane l16 supplies s = s0+l16; the lane half picks
  // the d sub-range (K=0..15 vs K=16..31 of the 32-wide d-chunk).
  const float* xrow = X + ((size_t)b * Sn + (size_t)(s0 + l16)) * Dn + 16 * half;

  v8f acc[4] = {v8f{}, v8f{}, v8f{}, v8f{}};

  #pragma unroll 2
  for (int d0 = 0; d0 < Dn; d0 += 32) {
    const v16bf bmat = cvt_bf16_16(xrow + d0);
    #pragma unroll
    for (int t = 0; t < 4; ++t) {
      const int k0 = (wave * 4 + t) * 16;
      // A-operand: lane l16 supplies row k0+l16; half selects the
      // {0..15} vs {16..31} d-element range (same as B).
      const float* wr = W + (size_t)(k0 + l16) * Dn + d0 + 16 * half;
      const v16bf amat = cvt_bf16_8_8(wr, wr + 8);
      acc[t] = __builtin_amdgcn_wmma_f32_16x16x32_bf16(
          false, amat, false, bmat, (short)0, acc[t], false, false);
    }
  }

  // C layout: VGPR m, lanes 0-15 -> (M=m, N=lane); lanes 16-31 -> (M=m+8, N=lane-16).
  // So this lane's accumulator element m corresponds to k = k0 + 8*half + m, s = s0+l16.
  const float* wgb = wg + (size_t)b * Kn;
  float part = 0.f;
  #pragma unroll
  for (int t = 0; t < 4; ++t) {
    const int kbase = (wave * 4 + t) * 16 + 8 * half;
    #pragma unroll
    for (int m = 0; m < 8; ++m)
      part += wgb[kbase + m] * tanhf(acc[t][m]);
  }
  // Merge the two lane halves (same N=s, complementary M sets).
  part += __shfl_xor(part, 16);

  __shared__ float red[4][16];
  if (half == 0) red[wave][l16] = part;
  __syncthreads();
  if (threadIdx.x < 16) {
    const float sum = red[0][threadIdx.x] + red[1][threadIdx.x] +
                      red[2][threadIdx.x] + red[3][threadIdx.x];
    scores[((size_t)which * Bn + b) * Sn + s0 + threadIdx.x] = sum;
  }
}

// ---------------------------------------------------------------------------
// Kernel 4: softmax over S (in LDS) + attention-weighted pooling.
// grid = (B, 2), block = 256. Thread t owns d = t and t+256 (coalesced).
// out layout: [vector_v (B*D) | vector_q (B*D)] flat f32.
// ---------------------------------------------------------------------------
__global__ __launch_bounds__(256) void softmax_pool_kernel(
    const float* __restrict__ Vn, const float* __restrict__ Qn,
    const float* __restrict__ scores, float* __restrict__ out) {
  const int b     = blockIdx.x;
  const int which = blockIdx.y;
  const int tid   = threadIdx.x;

  __shared__ float alpha[Sn];
  __shared__ float red[256];

  const float* sc = scores + ((size_t)which * Bn + b) * Sn;
  const float s0 = sc[tid], s1 = sc[tid + 256];

  // max reduce
  red[tid] = fmaxf(s0, s1);
  __syncthreads();
  for (int off = 128; off > 0; off >>= 1) {
    if (tid < off) red[tid] = fmaxf(red[tid], red[tid + off]);
    __syncthreads();
  }
  const float mx = red[0];
  __syncthreads();

  // exp + sum reduce
  const float e0 = __expf(s0 - mx), e1 = __expf(s1 - mx);
  alpha[tid] = e0;
  alpha[tid + 256] = e1;
  red[tid] = e0 + e1;
  __syncthreads();
  for (int off = 128; off > 0; off >>= 1) {
    if (tid < off) red[tid] += red[tid + off];
    __syncthreads();
  }
  const float inv = 1.0f / red[0];
  __syncthreads();
  alpha[tid]       *= inv;
  alpha[tid + 256] *= inv;
  __syncthreads();

  // weighted pooling over S
  const float* X = which ? Qn : Vn;
  const float* xb = X + (size_t)b * Sn * Dn;
  float acc0 = 0.f, acc1 = 0.f;
  for (int s = 0; s < Sn; ++s) {
    const float a = alpha[s];
    const size_t r = (size_t)s * Dn;
    acc0 += a * xb[r + tid];
    acc1 += a * xb[r + tid + 256];
  }
  out[((size_t)which * Bn + b) * Dn + tid]       = acc0;
  out[((size_t)which * Bn + b) * Dn + tid + 256] = acc1;
}

// ---------------------------------------------------------------------------
// Launch
// ---------------------------------------------------------------------------
extern "C" void kernel_launch(void* const* d_in, const int* in_sizes, int n_in,
                              void* d_out, int out_size, void* d_ws, size_t ws_size,
                              hipStream_t stream) {
  const float* Vn = (const float*)d_in[0];   // [B,S,D]
  const float* Qn = (const float*)d_in[1];   // [B,S,D]
  const float* Wv = (const float*)d_in[2];   // [K,D]
  const float* Wq = (const float*)d_in[3];   // [K,D]
  const float* Wm = (const float*)d_in[4];   // [K,D]
  const float* Wh = (const float*)d_in[5];   // [1,K]
  float* out = (float*)d_out;                // [2*B*D]

  // Workspace layout (f32): M0 [B*D] | wg [B*K] | scores [2*B*S]  (~458 KB)
  float* M0     = (float*)d_ws;
  float* wg     = M0 + (size_t)Bn * Dn;
  float* scores = wg + (size_t)Bn * Kn;

  mean_tanh_kernel<<<Bn, 256, 0, stream>>>(Vn, Qn, M0);
  gate_kernel<<<Bn, 256, 0, stream>>>(Wm, Wh, M0, wg);
  scores_kernel<<<dim3(Bn, Sn / 16, 2), 128, 0, stream>>>(Vn, Qn, Wv, Wq, wg, scores);
  softmax_pool_kernel<<<dim3(Bn, 2), 256, 0, stream>>>(Vn, Qn, scores, out);
}
